// SeparateParamsGroupedExperts_9912784519935
// MI455X (gfx1250) — matrix operand order repaired and checked by reading, not compiled, not run on hardware
//
#include <hip/hip_runtime.h>

// Grouped-expert SwiGLU MLP for MI455X (gfx1250, wave32, WMMA).
//   pre:  x, Wg, Wd, Wu  fp32 -> bf16  (one streaming pass, if ws allows)
//   h   = silu(x @ Wg) * (x @ Wd)      [E, T, D_H]  bf16 in d_ws
//   out = h @ Wu                        [E, T, D_IN] fp32 in d_out
// v_wmma_f32_16x16x32_bf16 core; double-buffered pipeline with
// global_load_async_to_lds_b128 staging for A-tiles (ASYNCcnt), register
// staging for transposed B-tiles; one barrier per K-step.

namespace {

constexpr int kE   = 8;
constexpr int kT   = 2048;
constexpr int kDin = 2048;
constexpr int kDh  = 5632;

constexpr int BM = 128;   // token rows per block
constexpr int BN = 64;    // output cols per block
constexpr int BK = 32;    // K step == WMMA K for bf16
constexpr int LD = 40;    // padded LDS row stride in bf16 elems (80B, 16B aligned)

typedef __attribute__((ext_vector_type(16))) __bf16 bf16x16;
typedef __attribute__((ext_vector_type(8)))  float  floatx8;
typedef __attribute__((ext_vector_type(4)))  int    v4i;
typedef __attribute__((address_space(1))) v4i v4i_as1;   // global
typedef __attribute__((address_space(3))) v4i v4i_as3;   // LDS

struct Bytes32 { uint4 lo, hi; };

#if defined(__gfx1250__) && __has_builtin(__builtin_amdgcn_global_load_async_to_lds_b128)
#define HAVE_ASYNC_LDS 1
#else
#define HAVE_ASYNC_LDS 0
#endif

#if HAVE_ASYNC_LDS
__device__ __forceinline__ void async_cp16(const void* g, void* l) {
  __builtin_amdgcn_global_load_async_to_lds_b128(
      (v4i_as1*)g, (v4i_as3*)l, /*offset=*/0, /*cpol=*/0);
}
__device__ __forceinline__ void wait_async_zero() {
#if __has_builtin(__builtin_amdgcn_s_wait_asynccnt)
  __builtin_amdgcn_s_wait_asynccnt(0);
#else
  asm volatile("s_wait_asynccnt 0x0" ::: "memory");
#endif
}
#endif

__device__ __forceinline__ unsigned short f2bf(float f) {
  unsigned int u = __builtin_bit_cast(unsigned int, f);
  u += 0x7FFFu + ((u >> 16) & 1u);   // round to nearest even
  return (unsigned short)(u >> 16);
}

__device__ __forceinline__ unsigned int pk2(float lo, float hi) {
  return (unsigned int)f2bf(lo) | ((unsigned int)f2bf(hi) << 16);
}

// Assemble a 16-element bf16 fragment from two 16-byte LDS runs.
__device__ __forceinline__ bf16x16 ld_frag(const unsigned short* p0,
                                           const unsigned short* p1) {
  Bytes32 t;
  t.lo = *(const uint4*)p0;
  t.hi = *(const uint4*)p1;
  return __builtin_bit_cast(bf16x16, t);
}

__device__ __forceinline__ floatx8 wmma_bf16(bf16x16 a, bf16x16 b, floatx8 c) {
  return __builtin_amdgcn_wmma_f32_16x16x32_bf16(
      /*neg_a=*/false, a, /*neg_b=*/false, b,
      /*c_mod=*/(short)0, c, /*reuse_a=*/false, /*reuse_b=*/false);
}

// ---- staging registers (unused fields are dead-code-eliminated) -----------
struct AStage { uint4 q[2]; float4 f[4]; };
struct BStage { uint4 q;    float4 f[2]; };

// A-tile: 32B (16 bf16) per thread, row major. bf16+async goes straight to
// LDS at issue time; other paths load regs at issue, store at commit.
template <bool BF16SRC>
__device__ __forceinline__ void issue_a(const void* base, size_t elemOff,
                                        unsigned short* ldsRow, AStage& st) {
  if constexpr (BF16SRC) {
    const unsigned short* src = (const unsigned short*)base + elemOff;
#if HAVE_ASYNC_LDS
    async_cp16(src, ldsRow);
    async_cp16(src + 8, ldsRow + 8);
    (void)st;
#else
    const uint4* s = (const uint4*)src;
    st.q[0] = s[0];
    st.q[1] = s[1];
    (void)ldsRow;
#endif
  } else {
    const float4* s = (const float4*)((const float*)base + elemOff);
    st.f[0] = s[0]; st.f[1] = s[1]; st.f[2] = s[2]; st.f[3] = s[3];
    (void)ldsRow;
  }
}

template <bool BF16SRC>
__device__ __forceinline__ void commit_a(const AStage& st,
                                         unsigned short* ldsRow) {
  if constexpr (BF16SRC) {
#if HAVE_ASYNC_LDS
    (void)st; (void)ldsRow;   // already in flight to LDS
#else
    *(uint4*)ldsRow       = st.q[0];
    *(uint4*)(ldsRow + 8) = st.q[1];
#endif
  } else {
    float4 f0 = st.f[0], f1 = st.f[1], f2 = st.f[2], f3 = st.f[3];
    *(uint4*)ldsRow = make_uint4(pk2(f0.x, f0.y), pk2(f0.z, f0.w),
                                 pk2(f1.x, f1.y), pk2(f1.z, f1.w));
    *(uint4*)(ldsRow + 8) = make_uint4(pk2(f2.x, f2.y), pk2(f2.z, f2.w),
                                       pk2(f3.x, f3.y), pk2(f3.z, f3.w));
  }
}

// B-tile: one K row x 8 N cols per thread; commit transposes into [N][K].
template <bool BF16SRC>
__device__ __forceinline__ void issue_b(const void* base, size_t elemOff,
                                        BStage& st) {
  if constexpr (BF16SRC) {
    st.q = *(const uint4*)((const unsigned short*)base + elemOff);
  } else {
    const float4* s = (const float4*)((const float*)base + elemOff);
    st.f[0] = s[0];
    st.f[1] = s[1];
  }
}

template <bool BF16SRC>
__device__ __forceinline__ void commit_b(const BStage& st, unsigned short* lds,
                                         int n8, int k) {
  if constexpr (BF16SRC) {
    uint4 v = st.q;
    lds[(n8 + 0) * LD + k] = (unsigned short)(v.x);
    lds[(n8 + 1) * LD + k] = (unsigned short)(v.x >> 16);
    lds[(n8 + 2) * LD + k] = (unsigned short)(v.y);
    lds[(n8 + 3) * LD + k] = (unsigned short)(v.y >> 16);
    lds[(n8 + 4) * LD + k] = (unsigned short)(v.z);
    lds[(n8 + 5) * LD + k] = (unsigned short)(v.z >> 16);
    lds[(n8 + 6) * LD + k] = (unsigned short)(v.w);
    lds[(n8 + 7) * LD + k] = (unsigned short)(v.w >> 16);
  } else {
    float4 f0 = st.f[0], f1 = st.f[1];
    lds[(n8 + 0) * LD + k] = f2bf(f0.x);
    lds[(n8 + 1) * LD + k] = f2bf(f0.y);
    lds[(n8 + 2) * LD + k] = f2bf(f0.z);
    lds[(n8 + 3) * LD + k] = f2bf(f0.w);
    lds[(n8 + 4) * LD + k] = f2bf(f1.x);
    lds[(n8 + 5) * LD + k] = f2bf(f1.y);
    lds[(n8 + 6) * LD + k] = f2bf(f1.z);
    lds[(n8 + 7) * LD + k] = f2bf(f1.w);
  }
}

// ---------------------------------------------------------------------------
// fp32 -> bf16 streaming convert (pre-pass). One float4 -> uint2 per thread.
// ---------------------------------------------------------------------------
__global__ __launch_bounds__(256) void k_cvt_bf16(
    const float4* __restrict__ src, uint2* __restrict__ dst, int n4) {
  int i = blockIdx.x * 256 + threadIdx.x;
  if (i < n4) {
    float4 f = src[i];
    dst[i] = make_uint2(pk2(f.x, f.y), pk2(f.z, f.w));
  }
}

// ---------------------------------------------------------------------------
// Kernel 1: h = silu(x@Wg) * (x@Wd), bf16 out.
// Grid: (kDh/BN, kT/BM, kE), 256 threads = 8 waves (4x2), double-buffered.
// ---------------------------------------------------------------------------
template <bool BF16SRC>
__global__ __launch_bounds__(256) void k_gate_up(
    const void* __restrict__ x, const void* __restrict__ wg,
    const void* __restrict__ wd, unsigned short* __restrict__ hbuf) {
  __shared__ __align__(16) unsigned short sX[2][BM * LD];
  __shared__ __align__(16) unsigned short sG[2][BN * LD];
  __shared__ __align__(16) unsigned short sU[2][BN * LD];

  const int e  = blockIdx.z;
  const int m0 = blockIdx.y * BM;
  const int n0 = blockIdx.x * BN;

  const size_t xOff = (size_t)e * kT * kDin;
  const size_t wOff = (size_t)e * kDin * kDh;
  const void* xe = BF16SRC ? (const void*)((const unsigned short*)x + xOff)
                           : (const void*)((const float*)x + xOff);
  const void* ge = BF16SRC ? (const void*)((const unsigned short*)wg + wOff)
                           : (const void*)((const float*)wg + wOff);
  const void* de = BF16SRC ? (const void*)((const unsigned short*)wd + wOff)
                           : (const void*)((const float*)wd + wOff);
  unsigned short* he = hbuf + (size_t)e * kT * kDh;

  const int tid  = threadIdx.x;
  const int lane = tid & 31;
  const int wave = tid >> 5;
  const int wm   = wave & 3;
  const int wn   = wave >> 2;
  const int l16  = lane & 15;
  const int kA   = (lane < 16) ? 0 : 16;  // A frag K base (contiguous 16 run)
  const int kB   = (lane < 16) ? 0 : 16;  // B frag K base (contiguous 16 run)

  const int xr  = tid >> 1;           // 0..127
  const int xc  = (tid & 1) * 16;     // 0 or 16
  const int wk  = tid >> 3;           // 0..31
  const int wn8 = (tid & 7) * 8;      // 0..56
  const int aLds = xr * LD + xc;

  floatx8 accG[2][2] = {};
  floatx8 accU[2][2] = {};

  AStage ax; BStage bgs, bus;

  // ---- prologue: fill tile 0 into buffer 0 ----
  issue_a<BF16SRC>(xe, (size_t)(m0 + xr) * kDin + xc, &sX[0][aLds], ax);
  issue_b<BF16SRC>(ge, (size_t)wk * kDh + n0 + wn8, bgs);
  issue_b<BF16SRC>(de, (size_t)wk * kDh + n0 + wn8, bus);
  commit_a<BF16SRC>(ax, &sX[0][aLds]);
  commit_b<BF16SRC>(bgs, sG[0], wn8, wk);
  commit_b<BF16SRC>(bus, sU[0], wn8, wk);
#if HAVE_ASYNC_LDS
  if constexpr (BF16SRC) wait_async_zero();
#endif
  __syncthreads();

  constexpr int nIter = kDin / BK;
  for (int it = 0; it < nIter; ++it) {
    const int cur = it & 1;
    const int nxt = cur ^ 1;
    const bool hasNext = (it + 1) < nIter;

    // prefetch tile it+1 (A async into alternate LDS buffer; B into regs)
    if (hasNext) {
      const int kk = (it + 1) * BK;
      issue_a<BF16SRC>(xe, (size_t)(m0 + xr) * kDin + kk + xc, &sX[nxt][aLds], ax);
      issue_b<BF16SRC>(ge, (size_t)(kk + wk) * kDh + n0 + wn8, bgs);
      issue_b<BF16SRC>(de, (size_t)(kk + wk) * kDh + n0 + wn8, bus);
    }

    // compute on current buffer
    const int ar0 = (wm * 32 + l16) * LD;
    const int ar1 = (wm * 32 + 16 + l16) * LD;
    bf16x16 a0 = ld_frag(&sX[cur][ar0 + kA], &sX[cur][ar0 + kA + 8]);
    bf16x16 a1 = ld_frag(&sX[cur][ar1 + kA], &sX[cur][ar1 + kA + 8]);

    const int bc0 = (wn * 32 + l16) * LD;
    const int bc1 = (wn * 32 + 16 + l16) * LD;
    bf16x16 bg0 = ld_frag(&sG[cur][bc0 + kB], &sG[cur][bc0 + kB + 8]);
    bf16x16 bg1 = ld_frag(&sG[cur][bc1 + kB], &sG[cur][bc1 + kB + 8]);
    bf16x16 bu0 = ld_frag(&sU[cur][bc0 + kB], &sU[cur][bc0 + kB + 8]);
    bf16x16 bu1 = ld_frag(&sU[cur][bc1 + kB], &sU[cur][bc1 + kB + 8]);

    accG[0][0] = wmma_bf16(a0, bg0, accG[0][0]);
    accG[0][1] = wmma_bf16(a0, bg1, accG[0][1]);
    accG[1][0] = wmma_bf16(a1, bg0, accG[1][0]);
    accG[1][1] = wmma_bf16(a1, bg1, accG[1][1]);
    accU[0][0] = wmma_bf16(a0, bu0, accU[0][0]);
    accU[0][1] = wmma_bf16(a0, bu1, accU[0][1]);
    accU[1][0] = wmma_bf16(a1, bu0, accU[1][0]);
    accU[1][1] = wmma_bf16(a1, bu1, accU[1][1]);

    // commit tile it+1 and synchronize (single barrier per K-step)
    if (hasNext) {
      commit_a<BF16SRC>(ax, &sX[nxt][aLds]);
      commit_b<BF16SRC>(bgs, sG[nxt], wn8, wk);
      commit_b<BF16SRC>(bus, sU[nxt], wn8, wk);
#if HAVE_ASYNC_LDS
      if constexpr (BF16SRC) wait_async_zero();
#endif
      __syncthreads();
    }
  }

  // epilogue: h = g * sigmoid(g) * u  (v_exp + v_rcp, no IEEE division)
  const int rofs = (lane < 16) ? 0 : 8;
#pragma unroll
  for (int i = 0; i < 2; ++i) {
#pragma unroll
    for (int j = 0; j < 2; ++j) {
      const int col = n0 + wn * 32 + j * 16 + l16;
      const int rb  = m0 + wm * 32 + i * 16 + rofs;
      floatx8 g = accG[i][j];
      floatx8 u = accU[i][j];
#pragma unroll
      for (int r = 0; r < 8; ++r) {
        float gv = g[r];
        float sg = __builtin_amdgcn_rcpf(1.0f + __expf(-gv));
        he[(size_t)(rb + r) * kDh + col] = f2bf(gv * sg * u[r]);
      }
    }
  }
}

// ---------------------------------------------------------------------------
// Kernel 2: out = h @ Wu, fp32 out. Grid: (kDin/BN, kT/BM, kE).
// ---------------------------------------------------------------------------
template <bool BF16SRC>
__global__ __launch_bounds__(256) void k_down(
    const unsigned short* __restrict__ hbuf, const void* __restrict__ wu,
    float* __restrict__ out) {
  __shared__ __align__(16) unsigned short sA[2][BM * LD];
  __shared__ __align__(16) unsigned short sB[2][BN * LD];

  const int e  = blockIdx.z;
  const int m0 = blockIdx.y * BM;
  const int n0 = blockIdx.x * BN;

  const unsigned short* he = hbuf + (size_t)e * kT * kDh;
  const size_t uOff = (size_t)e * kDh * kDin;
  const void* ue = BF16SRC ? (const void*)((const unsigned short*)wu + uOff)
                           : (const void*)((const float*)wu + uOff);
  float* oe = out + (size_t)e * kT * kDin;

  const int tid  = threadIdx.x;
  const int lane = tid & 31;
  const int wave = tid >> 5;
  const int wm   = wave & 3;
  const int wn   = wave >> 2;
  const int l16  = lane & 15;
  const int kA   = (lane < 16) ? 0 : 16;
  const int kB   = (lane < 16) ? 0 : 16;

  const int xr  = tid >> 1;
  const int xc  = (tid & 1) * 16;
  const int wk  = tid >> 3;
  const int wn8 = (tid & 7) * 8;
  const int aLds = xr * LD + xc;

  floatx8 acc[2][2] = {};

  AStage ax; BStage bs;

  issue_a<true>(he, (size_t)(m0 + xr) * kDh + xc, &sA[0][aLds], ax);
  issue_b<BF16SRC>(ue, (size_t)wk * kDin + n0 + wn8, bs);
  commit_a<true>(ax, &sA[0][aLds]);
  commit_b<BF16SRC>(bs, sB[0], wn8, wk);
#if HAVE_ASYNC_LDS
  wait_async_zero();
#endif
  __syncthreads();

  constexpr int nIter = kDh / BK;
  for (int it = 0; it < nIter; ++it) {
    const int cur = it & 1;
    const int nxt = cur ^ 1;
    const bool hasNext = (it + 1) < nIter;

    if (hasNext) {
      const int kk = (it + 1) * BK;
      issue_a<true>(he, (size_t)(m0 + xr) * kDh + kk + xc, &sA[nxt][aLds], ax);
      issue_b<BF16SRC>(ue, (size_t)(kk + wk) * kDin + n0 + wn8, bs);
    }

    const int ar0 = (wm * 32 + l16) * LD;
    const int ar1 = (wm * 32 + 16 + l16) * LD;
    bf16x16 a0 = ld_frag(&sA[cur][ar0 + kA], &sA[cur][ar0 + kA + 8]);
    bf16x16 a1 = ld_frag(&sA[cur][ar1 + kA], &sA[cur][ar1 + kA + 8]);

    const int bc0 = (wn * 32 + l16) * LD;
    const int bc1 = (wn * 32 + 16 + l16) * LD;
    bf16x16 b0 = ld_frag(&sB[cur][bc0 + kB], &sB[cur][bc0 + kB + 8]);
    bf16x16 b1 = ld_frag(&sB[cur][bc1 + kB], &sB[cur][bc1 + kB + 8]);

    acc[0][0] = wmma_bf16(a0, b0, acc[0][0]);
    acc[0][1] = wmma_bf16(a0, b1, acc[0][1]);
    acc[1][0] = wmma_bf16(a1, b0, acc[1][0]);
    acc[1][1] = wmma_bf16(a1, b1, acc[1][1]);

    if (hasNext) {
      commit_a<true>(ax, &sA[nxt][aLds]);
      commit_b<BF16SRC>(bs, sB[nxt], wn8, wk);
#if HAVE_ASYNC_LDS
      wait_async_zero();
#endif
      __syncthreads();
    }
  }

  const int rofs = (lane < 16) ? 0 : 8;
#pragma unroll
  for (int i = 0; i < 2; ++i) {
#pragma unroll
    for (int j = 0; j < 2; ++j) {
      const int col = n0 + wn * 32 + j * 16 + l16;
      const int rb  = m0 + wm * 32 + i * 16 + rofs;
      floatx8 c = acc[i][j];
#pragma unroll
      for (int r = 0; r < 8; ++r) {
        oe[(size_t)(rb + r) * kDin + col] = c[r];
      }
    }
  }
}

}  // namespace

extern "C" void kernel_launch(void* const* d_in, const int* in_sizes, int n_in,
                              void* d_out, int out_size, void* d_ws, size_t ws_size,
                              hipStream_t stream) {
  (void)in_sizes; (void)n_in; (void)out_size;

  const float* x  = (const float*)d_in[0];   // [E, T, D_IN]
  const float* wg = (const float*)d_in[1];   // [E, D_IN, D_H]
  const float* wd = (const float*)d_in[2];   // [E, D_IN, D_H]
  const float* wu = (const float*)d_in[3];   // [E, D_H, D_IN]

  const size_t nX = (size_t)kE * kT * kDin;       // 33.5M elems
  const size_t nW = (size_t)kE * kDin * kDh;      // 92.3M elems
  const size_t nH = (size_t)kE * kT * kDh;        // 92.3M elems
  const size_t needFull = 2 * (nH + nX + 3 * nW); // h + bf16 copies: ~805 MB

  unsigned short* hbuf = (unsigned short*)d_ws;   // [E, T, D_H] bf16
  float* out = (float*)d_out;

  dim3 block(256);
  dim3 g1(kDh / BN, kT / BM, kE);   // 88 x 16 x 8
  dim3 g2(kDin / BN, kT / BM, kE);  // 32 x 16 x 8

  if (ws_size >= needFull) {
    // bf16 pre-pass: halves GEMM-phase tile traffic, cleans inner loops.
    unsigned short* xb  = hbuf + nH;
    unsigned short* wgb = xb + nX;
    unsigned short* wdb = wgb + nW;
    unsigned short* wub = wdb + nW;

    const int n4x = (int)(nX / 4);
    const int n4w = (int)(nW / 4);
    k_cvt_bf16<<<(n4x + 255) / 256, block, 0, stream>>>(
        (const float4*)x, (uint2*)xb, n4x);
    k_cvt_bf16<<<(n4w + 255) / 256, block, 0, stream>>>(
        (const float4*)wg, (uint2*)wgb, n4w);
    k_cvt_bf16<<<(n4w + 255) / 256, block, 0, stream>>>(
        (const float4*)wd, (uint2*)wdb, n4w);
    k_cvt_bf16<<<(n4w + 255) / 256, block, 0, stream>>>(
        (const float4*)wu, (uint2*)wub, n4w);

    k_gate_up<true><<<g1, block, 0, stream>>>(xb, wgb, wdb, hbuf);
    k_down<true><<<g2, block, 0, stream>>>(hbuf, wub, out);
  } else {
    // Fallback: convert fp32 tiles on the fly while staging to LDS.
    k_gate_up<false><<<g1, block, 0, stream>>>(x, wg, wd, hbuf);
    k_down<false><<<g2, block, 0, stream>>>(hbuf, wu, out);
  }
}
